// NRUCell_1211180777756
// MI455X (gfx1250) — matrix operand order repaired and checked by reading, not compiled, not run on hardware
//
#include <hip/hip_runtime.h>
#include <math.h>

// ---------------- problem constants ----------------
#define BATCH   16384
#define INPUT_D 512
#define HID_D   1024
#define MEM_D   1152
#define SQ      48
#define CDIM    2688          // INPUT + HID + MEM
#define HMDIM   2176          // HID + MEM
#define NCAT    256           // padded cat-weight rows (196 used)
#define LN_EPS  1e-5f
#define P_EPS   1e-12f

// GEMM1: M=32 rows/WG, N=1024, K staged in blocks
#define KB1     672           // CDIM / 4
#define KSTEPS1 21            // KB1 / 32
// GEMM2: M=32 rows/WG, N=256, K staged in blocks
#define KB2     128           // HMDIM / 17
#define KSTEPS2 4             // KB2 / 32

// workspace layout (bytes)
#define HM_BYTES   ((size_t)BATCH * HMDIM * 2)          // bf16 hm = [h | m_prev]
#define WH_BYTES   ((size_t)HID_D * CDIM * 2)           // bf16 W_h
#define WCAT_BYTES ((size_t)NCAT * HMDIM * 2)           // bf16 W_cat (padded)

typedef __bf16 v8bf  __attribute__((ext_vector_type(8)));
typedef __bf16 v16bf __attribute__((ext_vector_type(16)));
typedef float  v8f   __attribute__((ext_vector_type(8)));
typedef int    v4i   __attribute__((ext_vector_type(4)));

union BFrag { v16bf v; v8bf h[2]; };

// async global->LDS path (CDNA5); graceful fallback with visible warning
#if defined(__AMDGCN__)
  #if __has_builtin(__builtin_amdgcn_global_load_async_to_lds_b128) && \
      __has_builtin(__builtin_amdgcn_s_wait_asynccnt)
    #define HAVE_ASYNC_LDS 1
  #else
    #define HAVE_ASYNC_LDS 0
    #warning "gfx1250 async-to-LDS builtins unavailable; using VGPR staging fallback"
  #endif
#else
  #define HAVE_ASYNC_LDS 0
#endif

#if HAVE_ASYNC_LDS
typedef __attribute__((address_space(1))) v4i* gas_v4i_ptr;
typedef __attribute__((address_space(3))) v4i* las_v4i_ptr;
__device__ __forceinline__ void async_b128_to_lds(const void* g, void* l) {
    __builtin_amdgcn_global_load_async_to_lds_b128(
        (gas_v4i_ptr)(void*)g, (las_v4i_ptr)l, 0, 0);
}
#endif

__device__ __forceinline__ v8f wmma_bf16(v16bf a, v16bf b, v8f c) {
    // D = A(16x32 bf16) * B(32x16 bf16) + C(16x16 f32)
    return __builtin_amdgcn_wmma_f32_16x16x32_bf16(
        false, a, false, b, (short)0, c, false, false);
}

// ---------------- prep kernels ----------------
__global__ __launch_bounds__(256) void k_cvt_f32_bf16(const float* __restrict__ src,
                                                      unsigned short* __restrict__ dst_u,
                                                      int n) {
    __bf16* dst = (__bf16*)dst_u;
    int i = blockIdx.x * 256 + threadIdx.x;
    if (i < n) dst[i] = (__bf16)src[i];
}

__global__ __launch_bounds__(256) void k_build_wcat(const float* __restrict__ wva,
                                                    const float* __restrict__ wvb,
                                                    const float* __restrict__ wa,
                                                    const float* __restrict__ wb,
                                                    unsigned short* __restrict__ wcat_u) {
    __bf16* wcat = (__bf16*)wcat_u;
    int i = blockIdx.x * 256 + threadIdx.x;
    if (i >= NCAT * HMDIM) return;
    int row = i / HMDIM, col = i - row * HMDIM;
    float v;
    if      (row < 96)  v = wva[row * HMDIM + col];
    else if (row < 192) v = wvb[(row - 96) * HMDIM + col];
    else if (row < 194) v = wa[(row - 192) * HMDIM + col];
    else if (row < 196) v = wb[(row - 194) * HMDIM + col];
    else                v = 0.0f;
    wcat[i] = (__bf16)v;
}

__global__ __launch_bounds__(256) void k_build_bcat(const float* __restrict__ bva,
                                                    const float* __restrict__ bvb,
                                                    const float* __restrict__ ba,
                                                    const float* __restrict__ bb,
                                                    float* __restrict__ bcat) {
    int i = threadIdx.x;
    float v;
    if      (i < 96)  v = bva[i];
    else if (i < 192) v = bvb[i - 96];
    else if (i < 194) v = ba[i - 192];
    else if (i < 196) v = bb[i - 194];
    else              v = 0.0f;
    bcat[i] = v;
}

__global__ __launch_bounds__(256) void k_mprev_to_hm(const float* __restrict__ mprev,
                                                     unsigned short* __restrict__ hm_u) {
    __bf16* hm = (__bf16*)hm_u;
    int i = blockIdx.x * 256 + threadIdx.x;
    if (i >= BATCH * MEM_D) return;
    int r = i / MEM_D, c = i - r * MEM_D;
    hm[(size_t)r * HMDIM + HID_D + c] = (__bf16)mprev[i];
}

// ---------------- GEMM1 + bias + LayerNorm + ReLU ----------------
// workgroup: 256 threads = 8 waves; 32 batch rows x 1024 cols
// per wave: 8 N-tiles x 2 M-tiles = 16 accumulators; each B fragment feeds 2 wmmas
__global__ __launch_bounds__(256) void k_gemm1_ln_relu(
    const float* __restrict__ inp, const float* __restrict__ hprev,
    const float* __restrict__ mprev, const float* __restrict__ bh,
    const float* __restrict__ lng, const float* __restrict__ lnb,
    const unsigned short* __restrict__ whb_u,
    float* __restrict__ hOut, unsigned short* __restrict__ hm_u)
{
    const __bf16* whb = (const __bf16*)whb_u;
    __bf16* hm = (__bf16*)hm_u;

    __shared__ __bf16 As[32 * KB1];           // 43 KB
    __shared__ float pSum[8][32], pSq[8][32];
    __shared__ float muS[32], rsS[32];

    const int tid  = threadIdx.x;
    const int wave = tid >> 5;
    const int lane = tid & 31;
    const int ln16 = lane & 15;
    const int hi   = lane >> 4;
    const int rowBase = blockIdx.x * 32;

    v8f acc[2][8] = {};

    for (int kb = 0; kb < CDIM; kb += KB1) {
        __syncthreads();
        // cooperative stage of A (concat input|h_prev|m_prev) -> bf16 LDS
        for (int idx = tid; idx < 32 * KB1; idx += 256) {
            int r = idx / KB1, k = idx - r * KB1;
            int gk = kb + k, grow = rowBase + r;
            float v;
            if (gk < INPUT_D)              v = inp[(size_t)grow * INPUT_D + gk];
            else if (gk < INPUT_D + HID_D) v = hprev[(size_t)grow * HID_D + (gk - INPUT_D)];
            else                           v = mprev[(size_t)grow * MEM_D + (gk - INPUT_D - HID_D)];
            As[idx] = (__bf16)v;
        }
        __syncthreads();

        for (int kk = 0; kk < KSTEPS1; ++kk) {
            const int ks = kk * 32;
            BFrag a0, a1;
            const __bf16* ar0 = &As[ln16 * KB1];
            const __bf16* ar1 = &As[(16 + ln16) * KB1];
            a0.h[0] = *(const v8bf*)(ar0 + ks + hi * 16);
            a0.h[1] = *(const v8bf*)(ar0 + ks + hi * 16 + 8);
            a1.h[0] = *(const v8bf*)(ar1 + ks + hi * 16);
            a1.h[1] = *(const v8bf*)(ar1 + ks + hi * 16 + 8);
            const int gk = kb + ks + hi * 16;
            // batch all B-fragment loads first (staggered waits, latency hiding)
            BFrag b[8];
#pragma unroll
            for (int t = 0; t < 8; ++t) {
                const __bf16* brow =
                    whb + (size_t)((wave * 8 + t) * 16 + ln16) * CDIM + gk;
                b[t].h[0] = *(const v8bf*)(brow);
                b[t].h[1] = *(const v8bf*)(brow + 8);
            }
#pragma unroll
            for (int t = 0; t < 8; ++t) {
                acc[0][t] = wmma_bf16(a0.v, b[t].v, acc[0][t]);
                acc[1][t] = wmma_bf16(a1.v, b[t].v, acc[1][t]);
            }
        }
    }

    // bias
#pragma unroll
    for (int t = 0; t < 8; ++t) {
        float bv = bh[(wave * 8 + t) * 16 + ln16];
#pragma unroll
        for (int mi = 0; mi < 2; ++mi)
#pragma unroll
            for (int v = 0; v < 8; ++v) acc[mi][t][v] += bv;
    }

    // deterministic LN stats: half-wave shfl tree + fixed-order wave sum
#pragma unroll
    for (int mi = 0; mi < 2; ++mi)
#pragma unroll
    for (int v = 0; v < 8; ++v) {
        float s = 0.f, q = 0.f;
#pragma unroll
        for (int t = 0; t < 8; ++t) { float x = acc[mi][t][v]; s += x; q += x * x; }
        for (int off = 8; off; off >>= 1) {
            s += __shfl_xor(s, off, 32);
            q += __shfl_xor(q, off, 32);
        }
        int r = mi * 16 + v + hi * 8;
        if (ln16 == 0) { pSum[wave][r] = s; pSq[wave][r] = q; }
    }
    __syncthreads();
    if (tid < 32) {
        float s = 0.f, q = 0.f;
        for (int w = 0; w < 8; ++w) { s += pSum[w][tid]; q += pSq[w][tid]; }
        float mu  = s * (1.0f / HID_D);
        float var = q * (1.0f / HID_D) - mu * mu;
        muS[tid] = mu;
        rsS[tid] = rsqrtf(var + LN_EPS);
    }
    __syncthreads();

    // normalize + relu; write h (f32) and bf16 into hm staging
#pragma unroll
    for (int t = 0; t < 8; ++t) {
        int n = (wave * 8 + t) * 16 + ln16;
        float g = lng[n], bb2 = lnb[n];
#pragma unroll
        for (int mi = 0; mi < 2; ++mi)
#pragma unroll
        for (int v = 0; v < 8; ++v) {
            int r = mi * 16 + v + hi * 8;
            int grow = rowBase + r;
            float y = (acc[mi][t][v] - muS[r]) * rsS[r] * g + bb2;
            y = fmaxf(y, 0.0f);
            hOut[(size_t)grow * HID_D + n] = y;
            hm[(size_t)grow * HMDIM + n] = (__bf16)y;
        }
    }
}

// ---------------- GEMM2 + outer-product p-norm memory update ----------------
// workgroup: 256 threads = 8 waves; 32 batch rows x 256 cols
// per wave: 2 N-tiles x 2 M-tiles = 4 accumulators
__global__ __launch_bounds__(256) void k_gemm2_mem(
    const unsigned short* __restrict__ hm_u,
    const unsigned short* __restrict__ wcat_u,
    const float* __restrict__ bcat,
    const float* __restrict__ mprev,
    float* __restrict__ mOut)
{
    const __bf16* hm   = (const __bf16*)hm_u;
    const __bf16* wcat = (const __bf16*)wcat_u;

    __shared__ __bf16 As[32 * KB2];      // 8 KB
    __shared__ float U[32 * NCAT];       // 32 KB
    __shared__ float nrm[32 * 4];
    __shared__ float scl[32 * 4];

    const int tid  = threadIdx.x;
    const int wave = tid >> 5;
    const int lane = tid & 31;
    const int ln16 = lane & 15;
    const int hi   = lane >> 4;
    const int rowBase = blockIdx.x * 32;

    v8f acc[2][2] = {};

    for (int kb = 0; kb < HMDIM; kb += KB2) {
        __syncthreads();
#if HAVE_ASYNC_LDS
        // async DMA of the 32 x KB2 bf16 tile straight into LDS (ASYNCcnt tracked)
        for (int cidx = tid; cidx < 32 * (KB2 / 8); cidx += 256) {
            int r = cidx >> 4, c = cidx & 15;   // KB2/8 == 16 chunks per row
            async_b128_to_lds(&hm[(size_t)(rowBase + r) * HMDIM + kb + c * 8],
                              &As[r * KB2 + c * 8]);
        }
        __builtin_amdgcn_s_wait_asynccnt(0);
#else
        for (int cidx = tid; cidx < 32 * (KB2 / 8); cidx += 256) {
            int r = cidx >> 4, c = cidx & 15;
            *(v8bf*)&As[r * KB2 + c * 8] =
                *(const v8bf*)&hm[(size_t)(rowBase + r) * HMDIM + kb + c * 8];
        }
#endif
        __syncthreads();

        for (int kk = 0; kk < KSTEPS2; ++kk) {
            const int ks = kk * 32;
            BFrag a0, a1;
            const __bf16* ar0 = &As[ln16 * KB2];
            const __bf16* ar1 = &As[(16 + ln16) * KB2];
            a0.h[0] = *(const v8bf*)(ar0 + ks + hi * 16);
            a0.h[1] = *(const v8bf*)(ar0 + ks + hi * 16 + 8);
            a1.h[0] = *(const v8bf*)(ar1 + ks + hi * 16);
            a1.h[1] = *(const v8bf*)(ar1 + ks + hi * 16 + 8);
            const int gk = kb + ks + hi * 16;
            BFrag b[2];
#pragma unroll
            for (int t = 0; t < 2; ++t) {
                const __bf16* brow =
                    wcat + (size_t)((wave * 2 + t) * 16 + ln16) * HMDIM + gk;
                b[t].h[0] = *(const v8bf*)(brow);
                b[t].h[1] = *(const v8bf*)(brow + 8);
            }
#pragma unroll
            for (int t = 0; t < 2; ++t) {
                acc[0][t] = wmma_bf16(a0.v, b[t].v, acc[0][t]);
                acc[1][t] = wmma_bf16(a1.v, b[t].v, acc[1][t]);
            }
        }
    }

    // U = acc + bias -> LDS
#pragma unroll
    for (int t = 0; t < 2; ++t) {
        int n = (wave * 2 + t) * 16 + ln16;
        float bv = bcat[n];
#pragma unroll
        for (int mi = 0; mi < 2; ++mi)
#pragma unroll
        for (int v = 0; v < 8; ++v) {
            int r = mi * 16 + v + hi * 8;
            U[r * NCAT + n] = acc[mi][t][v] + bv;
        }
    }
    __syncthreads();

    // p=5 norm sums per (row, gate, k); thread (r = tid>>3, c = tid&7)
    const int r = tid >> 3;
    const int c = tid & 7;
    const float* Ur = &U[r * NCAT];
    float pa0 = 0.f, pa1 = 0.f, pb0 = 0.f, pb1 = 0.f;
    for (int m = c; m < MEM_D; m += 8) {
        int i0 = m / SQ,  j0 = m - i0 * SQ;           // k=0: flat=m
        int f1 = MEM_D + m;
        int i1 = f1 / SQ, j1 = f1 - i1 * SQ;          // k=1
        float va0 = Ur[i0]      * Ur[SQ + j0];
        float va1 = Ur[i1]      * Ur[SQ + j1];
        float vb0 = Ur[96 + i0] * Ur[96 + SQ + j0];
        float vb1 = Ur[96 + i1] * Ur[96 + SQ + j1];
        float s;
        s = va0 * va0; pa0 += s * s * fabsf(va0);
        s = va1 * va1; pa1 += s * s * fabsf(va1);
        s = vb0 * vb0; pb0 += s * s * fabsf(vb0);
        s = vb1 * vb1; pb1 += s * s * fabsf(vb1);
    }
    // deterministic 8-lane group reduction
    for (int off = 4; off; off >>= 1) {
        pa0 += __shfl_xor(pa0, off, 32);
        pa1 += __shfl_xor(pa1, off, 32);
        pb0 += __shfl_xor(pb0, off, 32);
        pb1 += __shfl_xor(pb1, off, 32);
    }
    if (c == 0) {
        nrm[r * 4 + 0] = pa0;   // gate a, k=0
        nrm[r * 4 + 1] = pa1;   // gate a, k=1
        nrm[r * 4 + 2] = pb0;   // gate b, k=0
        nrm[r * 4 + 3] = pb1;   // gate b, k=1
    }
    __syncthreads();
    if (tid < 128) {
        int rr = tid >> 2, g = (tid >> 1) & 1, k = tid & 1;
        float denom = fmaxf(powf(nrm[tid], 0.2f), P_EPS);
        float coef  = U[rr * NCAT + 192 + g * 2 + k];  // alpha@192/193, beta@194/195
        scl[tid] = 0.5f * coef / denom;                // mean over K=2 folded in
    }
    __syncthreads();

    // m_out = m_prev + mean_k(alpha_k * v_a) - mean_k(beta_k * v_b)
    const int grow = rowBase + r;
    for (int m = c; m < MEM_D; m += 8) {
        int i0 = m / SQ,  j0 = m - i0 * SQ;
        int f1 = MEM_D + m;
        int i1 = f1 / SQ, j1 = f1 - i1 * SQ;
        float res = mprev[(size_t)grow * MEM_D + m];
        res += scl[r * 4 + 0] * (Ur[i0] * Ur[SQ + j0]);
        res += scl[r * 4 + 1] * (Ur[i1] * Ur[SQ + j1]);
        res -= scl[r * 4 + 2] * (Ur[96 + i0] * Ur[96 + SQ + j0]);
        res -= scl[r * 4 + 3] * (Ur[96 + i1] * Ur[96 + SQ + j1]);
        mOut[(size_t)grow * MEM_D + m] = res;
    }
}

// ---------------- launch ----------------
extern "C" void kernel_launch(void* const* d_in, const int* in_sizes, int n_in,
                              void* d_out, int out_size, void* d_ws, size_t ws_size,
                              hipStream_t stream) {
    const float* inp   = (const float*)d_in[0];
    const float* hprev = (const float*)d_in[1];
    const float* mprev = (const float*)d_in[2];
    const float* W_h   = (const float*)d_in[3];
    const float* b_h   = (const float*)d_in[4];
    const float* ln_g  = (const float*)d_in[5];
    const float* ln_b  = (const float*)d_in[6];
    const float* W_a   = (const float*)d_in[7];
    const float* b_a   = (const float*)d_in[8];
    const float* W_b   = (const float*)d_in[9];
    const float* b_b   = (const float*)d_in[10];
    const float* W_va  = (const float*)d_in[11];
    const float* b_va  = (const float*)d_in[12];
    const float* W_vb  = (const float*)d_in[13];
    const float* b_vb  = (const float*)d_in[14];

    char* ws = (char*)d_ws;
    unsigned short* hm_u   = (unsigned short*)(ws);
    unsigned short* whb_u  = (unsigned short*)(ws + HM_BYTES);
    unsigned short* wcat_u = (unsigned short*)(ws + HM_BYTES + WH_BYTES);
    float*          bcat   = (float*)        (ws + HM_BYTES + WH_BYTES + WCAT_BYTES);

    float* hOut = (float*)d_out;                               // (B, 1024)
    float* mOut = (float*)d_out + (size_t)BATCH * HID_D;       // (B, 1152)

    {
        int n = HID_D * CDIM;
        k_cvt_f32_bf16<<<(n + 255) / 256, 256, 0, stream>>>(W_h, whb_u, n);
    }
    {
        int n = NCAT * HMDIM;
        k_build_wcat<<<(n + 255) / 256, 256, 0, stream>>>(W_va, W_vb, W_a, W_b, wcat_u);
    }
    k_build_bcat<<<1, 256, 0, stream>>>(b_va, b_vb, b_a, b_b, bcat);
    {
        int n = BATCH * MEM_D;
        k_mprev_to_hm<<<(n + 255) / 256, 256, 0, stream>>>(mprev, hm_u);
    }

    k_gemm1_ln_relu<<<BATCH / 32, 256, 0, stream>>>(
        inp, hprev, mprev, b_h, ln_g, ln_b, whb_u, hOut, hm_u);

    k_gemm2_mem<<<BATCH / 32, 256, 0, stream>>>(hm_u, wcat_u, bcat, mprev, mOut);
}
